// CompressNetwork_71236327571798
// MI455X (gfx1250) — compile-verified, matrix operand order checked by reading, not run on hardware
//
#include <hip/hip_runtime.h>

// Grouped per-slot MLP, fused two-layer GEMM for MI455X (gfx1250).
// B=512, M=32, D_IN=1024, D_H=4096, D_OUT=1024, fp32 in memory.
//
// Fast path (needs ws >= 536 MB): pre-pass converts W1/W2 to bf16 transposed
// [M][N][K] (K-contiguous): every B fragment is 2x global_load_b128, zero
// in-loop conversion. Fallback: direct fp32 W loads + v_cvt_pk_bf16_f32.
// One workgroup = (slot m, 32-row batch tile).
//   LDS: h tile 32x4096 bf16 (256 KB) + staged X tile 32x1024 bf16 (64 KB)
//        = 320 KB, the full WGP LDS.
// K-loops use an explicit depth-1 software pipeline (ping-pong B buffers,
// manual unroll x2, #pragma unroll 1) so the compiler cannot over-prefetch
// and spill fragments to scratch.

#define BB    512
#define MS    32
#define DIN   1024
#define DH    4096
#define DOUT  1024

typedef __attribute__((ext_vector_type(16))) __bf16        v16bf;
typedef __attribute__((ext_vector_type(8)))  float         v8f;
typedef __attribute__((ext_vector_type(4)))  float         f4;
typedef __attribute__((ext_vector_type(4)))  unsigned int  u4;

union BFrag {
    unsigned int u[8];
    u4           q[2];
    v16bf        v;
};

__device__ __forceinline__ unsigned int cvt_pk_bf16(float lo, float hi) {
    unsigned int r;
    asm("v_cvt_pk_bf16_f32 %0, %1, %2" : "=v"(r) : "v"(lo), "v"(hi));
    return r;
}

__device__ __forceinline__ unsigned short cvt_bf16(float x) {
    unsigned int r;
    asm("v_cvt_pk_bf16_f32 %0, %1, %2" : "=v"(r) : "v"(x), "v"(0.0f));
    return (unsigned short)r;
}

#define WMMA_BF16(A, Bf, C) \
    __builtin_amdgcn_wmma_f32_16x16x32_bf16(false, (A).v, false, (Bf).v, (short)0, (C), false, false)

// Stage the 32x1024 fp32 X tile into LDS as bf16 (row stride DIN/2 u32).
__device__ __forceinline__ void stage_x_tile(const float* __restrict__ Xm,
                                             unsigned int* __restrict__ Xs32) {
#pragma unroll
    for (int i = 0; i < 16; ++i) {
        const int linear = i * 256 + (int)threadIdx.x;
        const int row    = linear >> 7;
        const int col4   = (linear & 127) * 4;
        const float* s = Xm + (size_t)row * (MS * DIN) + col4 * 2;
        f4 p0 = *(const f4*)(s);
        f4 p1 = *(const f4*)(s + 4);
        u4 d;
        d.x = cvt_pk_bf16(p0.x, p0.y);
        d.y = cvt_pk_bf16(p0.z, p0.w);
        d.z = cvt_pk_bf16(p1.x, p1.y);
        d.w = cvt_pk_bf16(p1.z, p1.w);
        *(u4*)(Xs32 + row * (DIN / 2) + col4) = d;
    }
}

// ---------------------------------------------------------------------------
// Pre-pass: convert+transpose weights  src fp32 [S][K][N] -> dst bf16 [S][N][K]
// ---------------------------------------------------------------------------
extern "C" __global__ __launch_bounds__(256)
void wt_transpose_cvt(const float* __restrict__ src, unsigned short* __restrict__ dst,
                      int K, int N) {
    __shared__ unsigned short T[64][68];
    const int s  = blockIdx.z;
    const int n0 = blockIdx.x * 64;
    const int k0 = blockIdx.y * 64;
    const int t  = threadIdx.x;

    const float* S = src + ((size_t)s * K + k0) * N + n0;
    const int cr = t >> 4;
    const int cc = (t & 15) * 4;
#pragma unroll
    for (int i = 0; i < 4; ++i) {
        const int row = cr + i * 16;
        f4 p = *(const f4*)(S + (size_t)row * N + cc);
        *(unsigned int*)&T[row][cc]     = cvt_pk_bf16(p.x, p.y);
        *(unsigned int*)&T[row][cc + 2] = cvt_pk_bf16(p.z, p.w);
    }
    __syncthreads();

    unsigned short* D = dst + ((size_t)s * N + n0) * K + k0;
    const int n  = t >> 2;
    const int sg = t & 3;
#pragma unroll
    for (int half = 0; half < 2; ++half) {
        const int kk = (sg + 4 * half) * 8;
        unsigned int w[4];
#pragma unroll
        for (int j = 0; j < 4; ++j)
            w[j] = (unsigned int)T[kk + 2 * j][n] |
                   ((unsigned int)T[kk + 2 * j + 1][n] << 16);
        *(u4*)(D + (size_t)n * K + kk) = *(const u4*)w;
    }
}

// ---------------------------------------------------------------------------
// Fast-path fused GEMM: bf16 weights (transposed), explicit K pipeline.
// ---------------------------------------------------------------------------
extern "C" __global__ __launch_bounds__(256, 2)
void compress_mlp_fused_t(const float*        __restrict__ X,     // [B][M][DIN] fp32
                          const unsigned int* __restrict__ W1t,   // [M][DH][DIN/2]
                          const float*        __restrict__ b1,
                          const unsigned int* __restrict__ W2t,   // [M][DOUT][DH/2]
                          const float*        __restrict__ b2,
                          float*              __restrict__ OUT) {
    extern __shared__ unsigned short Hs[];               // 32*DH bf16 (256 KB)
    unsigned int* Xs32 = (unsigned int*)(Hs + 32 * DH);  // 32*DIN bf16 (64 KB)

    const int lane  = threadIdx.x & 31;
    const int wave  = threadIdx.x >> 5;
    const int hi    = (lane >> 4) & 1;
    const int lr    = lane & 15;
    const int h8    = 8 * hi;
    const int btile = blockIdx.x;
    const int m     = blockIdx.y;

    const float* Xm  = X + (size_t)btile * 32 * (MS * DIN) + (size_t)m * DIN;
    const unsigned int* W1m = W1t + (size_t)m * DH * (DIN / 2);
    const unsigned int* W2m = W2t + (size_t)m * DOUT * (DH / 2);
    const float* b1m = b1 + (size_t)m * DH;
    const float* b2m = b2 + (size_t)m * DOUT;
    float*       Om  = OUT + (size_t)btile * 32 * (MS * DOUT) + (size_t)m * DOUT;

    stage_x_tile(Xm, Xs32);

    // -------- Phase 1: h = relu(X @ W1 + b1) -> LDS (bf16) --------
    for (int nc = 0; nc < DH; nc += 8 * 64) {
        __syncthreads();
        const int nbase = nc + wave * 64;
        v8f acc[2][4] = {};

        const unsigned int* wrow[4];
#pragma unroll
        for (int nt = 0; nt < 4; ++nt)
            wrow[nt] = W1m + (size_t)(nbase + nt * 16 + lr) * (DIN / 2) + 4 * hi;
        const unsigned int* xrow0 = Xs32 + (size_t)lr * (DIN / 2) + 4 * hi;
        const unsigned int* xrow1 = xrow0 + 16 * (DIN / 2);

        BFrag bufA[4], bufB[4];
#pragma unroll
        for (int nt = 0; nt < 4; ++nt) {        // preload k=0
            bufA[nt].q[0] = *(const u4*)(wrow[nt]);
            bufA[nt].q[1] = *(const u4*)(wrow[nt] + 8);
        }

#pragma unroll 1
        for (int k0 = 0; k0 < DIN; k0 += 64) {
            {   // half 1: preload k0+32 into bufB, consume bufA (k0)
                const int kp = (k0 + 32) >> 1;
                const int kc = k0 >> 1;
#pragma unroll
                for (int nt = 0; nt < 4; ++nt) {
                    bufB[nt].q[0] = *(const u4*)(wrow[nt] + kp);
                    bufB[nt].q[1] = *(const u4*)(wrow[nt] + kp + 8);
                }
                BFrag a0, a1;
                a0.q[0] = *(const u4*)(xrow0 + kc);
                a0.q[1] = *(const u4*)(xrow0 + kc + 8);
                a1.q[0] = *(const u4*)(xrow1 + kc);
                a1.q[1] = *(const u4*)(xrow1 + kc + 8);
#pragma unroll
                for (int nt = 0; nt < 4; ++nt) {
                    acc[0][nt] = WMMA_BF16(a0, bufA[nt], acc[0][nt]);
                    acc[1][nt] = WMMA_BF16(a1, bufA[nt], acc[1][nt]);
                }
            }
            {   // half 2: preload k0+64 (or wrap) into bufA, consume bufB (k0+32)
                const int kn = (k0 + 64 < DIN) ? (k0 + 64) : 0;
                const int kp = kn >> 1;
                const int kc = (k0 + 32) >> 1;
#pragma unroll
                for (int nt = 0; nt < 4; ++nt) {
                    bufA[nt].q[0] = *(const u4*)(wrow[nt] + kp);
                    bufA[nt].q[1] = *(const u4*)(wrow[nt] + kp + 8);
                }
                BFrag a0, a1;
                a0.q[0] = *(const u4*)(xrow0 + kc);
                a0.q[1] = *(const u4*)(xrow0 + kc + 8);
                a1.q[0] = *(const u4*)(xrow1 + kc);
                a1.q[1] = *(const u4*)(xrow1 + kc + 8);
#pragma unroll
                for (int nt = 0; nt < 4; ++nt) {
                    acc[0][nt] = WMMA_BF16(a0, bufB[nt], acc[0][nt]);
                    acc[1][nt] = WMMA_BF16(a1, bufB[nt], acc[1][nt]);
                }
            }
        }

#pragma unroll
        for (int nt = 0; nt < 4; ++nt) {
            const int   col  = nbase + nt * 16 + lr;
            const float bias = b1m[col];
#pragma unroll
            for (int mt = 0; mt < 2; ++mt)
#pragma unroll
                for (int r = 0; r < 8; ++r) {
                    const int row = mt * 16 + h8 + r;
                    float vv = acc[mt][nt][r] + bias;
                    vv = vv > 0.0f ? vv : 0.0f;
                    Hs[(size_t)row * DH + col] = cvt_bf16(vv);
                }
        }
    }

    __syncthreads();

    // -------- Phase 2: out = h @ W2 + b2 --------
    const unsigned int* Hs32 = (const unsigned int*)Hs;
    for (int nc = 0; nc < DOUT; nc += 8 * 64) {
        const int nbase = nc + wave * 64;
        v8f acc[2][4] = {};

        const unsigned int* wrow[4];
#pragma unroll
        for (int nt = 0; nt < 4; ++nt)
            wrow[nt] = W2m + (size_t)(nbase + nt * 16 + lr) * (DH / 2) + 4 * hi;
        const unsigned int* hrow0 = Hs32 + (size_t)lr * (DH / 2) + 4 * hi;
        const unsigned int* hrow1 = hrow0 + 16 * (DH / 2);

        BFrag bufA[4], bufB[4];
#pragma unroll
        for (int nt = 0; nt < 4; ++nt) {
            bufA[nt].q[0] = *(const u4*)(wrow[nt]);
            bufA[nt].q[1] = *(const u4*)(wrow[nt] + 8);
        }

#pragma unroll 1
        for (int k0 = 0; k0 < DH; k0 += 64) {
            {
                const int kp = (k0 + 32) >> 1;
                const int kc = k0 >> 1;
#pragma unroll
                for (int nt = 0; nt < 4; ++nt) {
                    bufB[nt].q[0] = *(const u4*)(wrow[nt] + kp);
                    bufB[nt].q[1] = *(const u4*)(wrow[nt] + kp + 8);
                }
                BFrag a0, a1;
                a0.q[0] = *(const u4*)(hrow0 + kc);
                a0.q[1] = *(const u4*)(hrow0 + kc + 8);
                a1.q[0] = *(const u4*)(hrow1 + kc);
                a1.q[1] = *(const u4*)(hrow1 + kc + 8);
#pragma unroll
                for (int nt = 0; nt < 4; ++nt) {
                    acc[0][nt] = WMMA_BF16(a0, bufA[nt], acc[0][nt]);
                    acc[1][nt] = WMMA_BF16(a1, bufA[nt], acc[1][nt]);
                }
            }
            {
                const int kn = (k0 + 64 < DH) ? (k0 + 64) : 0;
                const int kp = kn >> 1;
                const int kc = (k0 + 32) >> 1;
#pragma unroll
                for (int nt = 0; nt < 4; ++nt) {
                    bufA[nt].q[0] = *(const u4*)(wrow[nt] + kp);
                    bufA[nt].q[1] = *(const u4*)(wrow[nt] + kp + 8);
                }
                BFrag a0, a1;
                a0.q[0] = *(const u4*)(hrow0 + kc);
                a0.q[1] = *(const u4*)(hrow0 + kc + 8);
                a1.q[0] = *(const u4*)(hrow1 + kc);
                a1.q[1] = *(const u4*)(hrow1 + kc + 8);
#pragma unroll
                for (int nt = 0; nt < 4; ++nt) {
                    acc[0][nt] = WMMA_BF16(a0, bufB[nt], acc[0][nt]);
                    acc[1][nt] = WMMA_BF16(a1, bufB[nt], acc[1][nt]);
                }
            }
        }

#pragma unroll
        for (int nt = 0; nt < 4; ++nt) {
            const int   col  = nbase + nt * 16 + lr;
            const float bias = b2m[col];
#pragma unroll
            for (int mt = 0; mt < 2; ++mt)
#pragma unroll
                for (int r = 0; r < 8; ++r) {
                    const int row = mt * 16 + h8 + r;
                    Om[(size_t)row * (MS * DOUT) + col] = acc[mt][nt][r] + bias;
                }
        }
        __syncthreads();
    }
}

// ---------------------------------------------------------------------------
// Fallback fused GEMM: fp32 weights, in-loop v_cvt_pk_bf16_f32.
// ---------------------------------------------------------------------------
extern "C" __global__ __launch_bounds__(256, 2)
void compress_mlp_fused(const float* __restrict__ X,
                        const float* __restrict__ W1,
                        const float* __restrict__ b1,
                        const float* __restrict__ W2,
                        const float* __restrict__ b2,
                        float* __restrict__ OUT) {
    extern __shared__ unsigned short Hs[];
    unsigned int* Xs32 = (unsigned int*)(Hs + 32 * DH);

    const int lane  = threadIdx.x & 31;
    const int wave  = threadIdx.x >> 5;
    const int hi    = (lane >> 4) & 1;
    const int lr    = lane & 15;
    const int h8    = 8 * hi;
    const int btile = blockIdx.x;
    const int m     = blockIdx.y;

    const float* Xm  = X  + (size_t)btile * 32 * (MS * DIN) + (size_t)m * DIN;
    const float* W1m = W1 + (size_t)m * DIN * DH;
    const float* W2m = W2 + (size_t)m * DH * DOUT;
    const float* b1m = b1 + (size_t)m * DH;
    const float* b2m = b2 + (size_t)m * DOUT;
    float*       Om  = OUT + (size_t)btile * 32 * (MS * DOUT) + (size_t)m * DOUT;

    stage_x_tile(Xm, Xs32);

    for (int nc = 0; nc < DH; nc += 8 * 64) {
        __syncthreads();
        const int nbase = nc + wave * 64;
        v8f acc[2][4] = {};

#pragma unroll 1
        for (int k0 = 0; k0 < DIN; k0 += 32) {
            BFrag a[2];
#pragma unroll
            for (int mt = 0; mt < 2; ++mt) {
                const unsigned int* xr =
                    Xs32 + (size_t)(mt * 16 + lr) * (DIN / 2) + (k0 >> 1) + 4 * hi;
                a[mt].q[0] = *(const u4*)(xr);
                a[mt].q[1] = *(const u4*)(xr + 8);
            }
#pragma unroll
            for (int nt = 0; nt < 4; ++nt) {
                const float* wr = W1m + (size_t)k0 * DH + (nbase + nt * 16 + lr);
                BFrag bfr;
#pragma unroll
                for (int v = 0; v < 8; ++v) {
                    const int ko = 2 * (v & 3) + 16 * (v >> 2) + h8;
                    bfr.u[v] = cvt_pk_bf16(wr[(size_t)ko * DH], wr[(size_t)(ko + 1) * DH]);
                }
                acc[0][nt] = WMMA_BF16(a[0], bfr, acc[0][nt]);
                acc[1][nt] = WMMA_BF16(a[1], bfr, acc[1][nt]);
            }
        }

#pragma unroll
        for (int nt = 0; nt < 4; ++nt) {
            const int   col  = nbase + nt * 16 + lr;
            const float bias = b1m[col];
#pragma unroll
            for (int mt = 0; mt < 2; ++mt)
#pragma unroll
                for (int r = 0; r < 8; ++r) {
                    const int row = mt * 16 + h8 + r;
                    float vv = acc[mt][nt][r] + bias;
                    vv = vv > 0.0f ? vv : 0.0f;
                    Hs[(size_t)row * DH + col] = cvt_bf16(vv);
                }
        }
    }

    __syncthreads();

    const unsigned int* Hs32 = (const unsigned int*)Hs;
    for (int nc = 0; nc < DOUT; nc += 8 * 64) {
        const int nbase = nc + wave * 64;
        v8f acc[2][4] = {};

#pragma unroll 1
        for (int k0 = 0; k0 < DH; k0 += 32) {
            BFrag a[2];
#pragma unroll
            for (int mt = 0; mt < 2; ++mt) {
                const unsigned int* hr =
                    Hs32 + (size_t)(mt * 16 + lr) * (DH / 2) + (k0 >> 1) + 4 * hi;
                a[mt].q[0] = *(const u4*)(hr);
                a[mt].q[1] = *(const u4*)(hr + 8);
            }
#pragma unroll
            for (int nt = 0; nt < 4; ++nt) {
                const float* wr = W2m + (size_t)k0 * DOUT + (nbase + nt * 16 + lr);
                BFrag bfr;
#pragma unroll
                for (int v = 0; v < 8; ++v) {
                    const int ko = 2 * (v & 3) + 16 * (v >> 2) + h8;
                    bfr.u[v] = cvt_pk_bf16(wr[(size_t)ko * DOUT], wr[(size_t)(ko + 1) * DOUT]);
                }
                acc[0][nt] = WMMA_BF16(a[0], bfr, acc[0][nt]);
                acc[1][nt] = WMMA_BF16(a[1], bfr, acc[1][nt]);
            }
        }

#pragma unroll
        for (int nt = 0; nt < 4; ++nt) {
            const int   col  = nbase + nt * 16 + lr;
            const float bias = b2m[col];
#pragma unroll
            for (int mt = 0; mt < 2; ++mt)
#pragma unroll
                for (int r = 0; r < 8; ++r) {
                    const int row = mt * 16 + h8 + r;
                    Om[(size_t)row * (MS * DOUT) + col] = acc[mt][nt][r] + bias;
                }
        }
        __syncthreads();
    }
}

// ---------------------------------------------------------------------------
extern "C" void kernel_launch(void* const* d_in, const int* in_sizes, int n_in,
                              void* d_out, int out_size, void* d_ws, size_t ws_size,
                              hipStream_t stream) {
    (void)in_sizes; (void)n_in; (void)out_size;

    const float* X  = (const float*)d_in[0];
    const float* W1 = (const float*)d_in[1];
    const float* b1 = (const float*)d_in[2];
    const float* W2 = (const float*)d_in[3];
    const float* b2 = (const float*)d_in[4];
    float* OUT = (float*)d_out;

    const size_t lds_bytes = (size_t)32 * DH * 2 + (size_t)32 * DIN * 2;  // 320 KB
    (void)hipFuncSetAttribute((const void*)compress_mlp_fused,
                              hipFuncAttributeMaxDynamicSharedMemorySize, (int)lds_bytes);
    (void)hipFuncSetAttribute((const void*)compress_mlp_fused_t,
                              hipFuncAttributeMaxDynamicSharedMemorySize, (int)lds_bytes);

    dim3 grid(BB / 32, MS);
    dim3 block(256);

    const size_t W1T_BYTES = (size_t)MS * DIN * DH * 2;      // 268 MB
    const size_t W2T_BYTES = (size_t)MS * DH * DOUT * 2;     // 268 MB
    const size_t NEED      = W1T_BYTES + W2T_BYTES;          // 536 MB

    if (ws_size >= NEED) {
        unsigned short* W1t = (unsigned short*)d_ws;
        unsigned short* W2t = (unsigned short*)((char*)d_ws + W1T_BYTES);

        dim3 g1(DH / 64, DIN / 64, MS);
        wt_transpose_cvt<<<g1, block, 0, stream>>>(W1, W1t, DIN, DH);
        dim3 g2(DOUT / 64, DH / 64, MS);
        wt_transpose_cvt<<<g2, block, 0, stream>>>(W2, W2t, DH, DOUT);

        compress_mlp_fused_t<<<grid, block, lds_bytes, stream>>>(
            X, (const unsigned int*)W1t, b1, (const unsigned int*)W2t, b2, OUT);
    } else {
        compress_mlp_fused<<<grid, block, lds_bytes, stream>>>(X, W1, b1, W2, b2, OUT);
    }
}
